// VanillaLSTM_57526791963079
// MI455X (gfx1250) — matrix operand order repaired and checked by reading, not compiled, not run on hardware
//
#include <hip/hip_runtime.h>

typedef __attribute__((ext_vector_type(16))) _Float16     v16h;
typedef __attribute__((ext_vector_type(8)))  float        v8f;
typedef __attribute__((ext_vector_type(4)))  unsigned int u32x4;

#define HD      128
#define NGATE   512
#define TENC    20
#define TDEC    25
#define SH_STRIDE  136   // h row stride in halves (128 + 8 pad, keeps 16B align)
#define SHF_STRIDE 132   // f32 h row stride
#define PACK_MAT_ELEMS (32 * 4 * 32 * 16)   // 65536 halves per 512x128 matrix

// ---------------------------------------------------------------------------
// Pre-pack: f32 (512x128) weights -> f16 fragments in WMMA B-matrix lane order
//   B fragment (K=32 x N=16): lane l -> N = l%16, K block = (l/16)*16, elems e=0..15 -> K
//   flat index: ((ntile*4 + kstep)*32 + lane)*16 + e
// ---------------------------------------------------------------------------
__global__ void pack_weights(const float* __restrict__ w0, const float* __restrict__ w1,
                             const float* __restrict__ w2, const float* __restrict__ w3,
                             const float* __restrict__ w4, const float* __restrict__ w5,
                             _Float16* __restrict__ packed) {
  int gid = blockIdx.x * blockDim.x + threadIdx.x;
  if (gid >= 6 * PACK_MAT_ELEMS) return;
  int mat   = gid >> 16;
  int pid   = gid & 65535;
  int e     = pid & 15;
  int lane  = (pid >> 4) & 31;
  int kstep = (pid >> 9) & 3;
  int ntile = pid >> 11;
  int n = ntile * 16 + (lane & 15);
  int k = kstep * 32 + ((lane >> 4) << 4) + e;
  const float* w = (mat == 0) ? w0 : (mat == 1) ? w1 : (mat == 2) ? w2
                  : (mat == 3) ? w3 : (mat == 4) ? w4 : w5;
  packed[gid] = (_Float16)w[n * HD + k];
}

__global__ void pack_bias(const float* __restrict__ bi0, const float* __restrict__ bh0,
                          const float* __restrict__ bi1, const float* __restrict__ bh1,
                          const float* __restrict__ bi2, const float* __restrict__ bh2,
                          const float* __restrict__ bi3, const float* __restrict__ bh3,
                          float* __restrict__ bsum) {
  int gid = blockIdx.x * blockDim.x + threadIdx.x;
  if (gid >= 4 * NGATE) return;
  int cell = gid >> 9, j = gid & 511;
  const float* a = cell == 0 ? bi0 : cell == 1 ? bi1 : cell == 2 ? bi2 : bi3;
  const float* b = cell == 0 ? bh0 : cell == 1 ? bh1 : cell == 2 ? bh2 : bh3;
  bsum[gid] = a[j] + b[j];
}

// ---------------------------------------------------------------------------
// Main persistent LSTM kernel
// ---------------------------------------------------------------------------
__device__ __forceinline__ float sigm(float x) { return 1.0f / (1.0f + __expf(-x)); }
__device__ __forceinline__ float tanh_fast(float x) {
  float e = __expf(-2.0f * x);
  return (1.0f - e) / (1.0f + e);
}

// A fragment (16x32 f16): lanes 0-15 M=lane, K = kb+[0..15];
// lanes 16-31 M=lane-16, K = kb+16+[0..15]
__device__ __forceinline__ v16h load_afrag(const _Float16* sh, int lane, int kb) {
  const int mA   = lane & 15;
  const int koff = (lane >> 4) << 4;
  const _Float16* p = sh + mA * SH_STRIDE + kb + koff;
  union { v16h v; u32x4 u[2]; } r;
  r.u[0] = *(const u32x4*)(p);
  r.u[1] = *(const u32x4*)(p + 8);
  return r.v;
}

__device__ __forceinline__ v16h load_bfrag(const _Float16* packed,
                                           int ntile, int kstep, int lane) {
  const _Float16* p = packed + (size_t)((((ntile << 2) + kstep) << 5) + lane) * 16;
  union { v16h v; u32x4 u[2]; } r;
  r.u[0] = ((const u32x4*)p)[0];
  r.u[1] = ((const u32x4*)p)[1];
  return r.v;
}

// acc[g] += shA(16x128) @ W^T slice for gate g, cols [wave*16, wave*16+16)
__device__ __forceinline__ void mm_acc(v8f acc[4], const _Float16* packedMat,
                                       const _Float16* shA, int wave, int lane) {
#pragma unroll
  for (int ks = 0; ks < 4; ++ks) {
    v16h a = load_afrag(shA, lane, ks * 32);
#pragma unroll
    for (int g = 0; g < 4; ++g) {
      v16h b = load_bfrag(packedMat, g * 8 + wave, ks, lane);
      acc[g] = __builtin_amdgcn_wmma_f32_16x16x32_f16(
          false, a, false, b, (short)0, acc[g], false, false);
    }
  }
}

__device__ __forceinline__ void state_update(v8f acc[4], v8f& c, _Float16* shH,
                                             float* shHf, int wave, int lane) {
  const int n = lane & 15;
  const int mbase = (lane >> 4) << 3;   // D layout: lanes 16-31 hold rows M+8
#pragma unroll
  for (int r = 0; r < 8; ++r) {
    float ig = sigm(acc[0][r]);
    float fg = sigm(acc[1][r]);
    float gg = tanh_fast(acc[2][r]);
    float og = sigm(acc[3][r]);
    float cn = fg * c[r] + ig * gg;
    c[r] = cn;
    float h = og * tanh_fast(cn);
    int m = mbase + r;
    shH[m * SH_STRIDE + wave * 16 + n] = (_Float16)h;
    if (shHf) shHf[m * SHF_STRIDE + wave * 16 + n] = h;
  }
}

__launch_bounds__(256)
__global__ void lstm_seq2seq(const float* __restrict__ target,
                             const _Float16* __restrict__ packed,
                             const float* __restrict__ bsum,
                             const float* __restrict__ encWih0,  // (512,7)
                             const float* __restrict__ decWih0,  // (512,2)
                             const float* __restrict__ outW,     // (2,128)
                             const float* __restrict__ outB,     // (2)
                             float* __restrict__ out) {          // (B,25,2)
  __shared__ __align__(16) _Float16 sh_h0[16 * SH_STRIDE];
  __shared__ __align__(16) _Float16 sh_h1[16 * SH_STRIDE];
  __shared__ __align__(16) float    sh_h1f[16 * SHF_STRIDE];
  __shared__ __align__(16) float    sh_x[16 * 8];
  __shared__ __align__(16) float    sh_outW[2 * HD];

  const int tid  = threadIdx.x;
  const int lane = tid & 31;
  const int wave = tid >> 5;
  const int m0   = blockIdx.x * 16;
  const int n     = lane & 15;
  const int mbase = (lane >> 4) << 3;

  for (int i = tid; i < 16 * SH_STRIDE; i += 256) {
    sh_h0[i] = (_Float16)0.0f;
    sh_h1[i] = (_Float16)0.0f;
  }
  if (tid < 2 * HD) sh_outW[tid] = outW[tid];

  // per-lane small weights / fused biases (lane owns gate cols g*128 + wave*16 + n)
  const int ng_base = wave * 16 + n;
  float wih0r[4][7], wdec0r[4][2];
  float bE0[4], bE1[4], bD0[4], bD1[4];
#pragma unroll
  for (int g = 0; g < 4; ++g) {
    int ng = g * 128 + ng_base;
#pragma unroll
    for (int k = 0; k < 7; ++k) wih0r[g][k] = encWih0[ng * 7 + k];
    wdec0r[g][0] = decWih0[ng * 2 + 0];
    wdec0r[g][1] = decWih0[ng * 2 + 1];
    bE0[g] = bsum[0 * NGATE + ng];
    bE1[g] = bsum[1 * NGATE + ng];
    bD0[g] = bsum[2 * NGATE + ng];
    bD1[g] = bsum[3 * NGATE + ng];
  }

  v8f c0 = {0.f, 0.f, 0.f, 0.f, 0.f, 0.f, 0.f, 0.f};
  v8f c1 = {0.f, 0.f, 0.f, 0.f, 0.f, 0.f, 0.f, 0.f};

  // ---------------- encoder: 2 fused stacked layers over 20 steps ----------
  for (int t = 0; t < TENC; ++t) {
    // Opaque zero byte-offset: defeats LICM (loads stay in-loop, no spills)
    // while keeping the base pointer's global address space (global_load,
    // LOADcnt only — decoupled from the LDS/DScnt stream).
    size_t woff = 0;
    asm volatile("" : "+s"(woff));
    const _Float16* pWhh0 = packed + woff + 0 * PACK_MAT_ELEMS;
    const _Float16* pWih1 = packed + woff + 1 * PACK_MAT_ELEMS;
    const _Float16* pWhh1 = packed + woff + 2 * PACK_MAT_ELEMS;

    if (tid < 112) {
      int row = tid / 7, col = tid % 7;
      sh_x[row * 8 + col] = target[((size_t)(m0 + row) * TENC + t) * 7 + col];
    }
    __syncthreads();

    v8f acc[4];
    // layer 0: bias + x@Wih0^T (K=7, VALU) + h0@Whh0^T (WMMA)
#pragma unroll
    for (int g = 0; g < 4; ++g)
#pragma unroll
      for (int r = 0; r < 8; ++r) acc[g][r] = bE0[g];
#pragma unroll
    for (int r = 0; r < 8; ++r) {
      int m = mbase + r;
#pragma unroll
      for (int k = 0; k < 7; ++k) {
        float xv = sh_x[m * 8 + k];
#pragma unroll
        for (int g = 0; g < 4; ++g) acc[g][r] += xv * wih0r[g][k];
      }
    }
    mm_acc(acc, pWhh0, sh_h0, wave, lane);
    __syncthreads();                       // all reads of old sh_h0 done
    state_update(acc, c0, sh_h0, nullptr, wave, lane);
    __syncthreads();                       // new sh_h0 visible

    // layer 1: bias + h0@Wih1^T + h1@Whh1^T (both WMMA)
#pragma unroll
    for (int g = 0; g < 4; ++g)
#pragma unroll
      for (int r = 0; r < 8; ++r) acc[g][r] = bE1[g];
    mm_acc(acc, pWih1, sh_h0, wave, lane);
    mm_acc(acc, pWhh1, sh_h1, wave, lane);
    __syncthreads();
    state_update(acc, c1, sh_h1, nullptr, wave, lane);
    __syncthreads();
  }

  // ---------------- decoder: autoregressive, 25 steps ----------------------
  if (tid < 32) sh_x[(tid >> 1) * 8 + (tid & 1)] = 0.0f;   // x0 = zeros
  __syncthreads();

  for (int t = 0; t < TDEC; ++t) {
    size_t woff = 0;
    asm volatile("" : "+s"(woff));
    const _Float16* pDWhh0 = packed + woff + 3 * PACK_MAT_ELEMS;
    const _Float16* pDWih1 = packed + woff + 4 * PACK_MAT_ELEMS;
    const _Float16* pDWhh1 = packed + woff + 5 * PACK_MAT_ELEMS;

    v8f acc[4];
    // dec layer 0: bias + pred@Wih^T (K=2, VALU) + h0@Whh^T (WMMA)
#pragma unroll
    for (int g = 0; g < 4; ++g)
#pragma unroll
      for (int r = 0; r < 8; ++r) acc[g][r] = bD0[g];
#pragma unroll
    for (int r = 0; r < 8; ++r) {
      int m = mbase + r;
#pragma unroll
      for (int k = 0; k < 2; ++k) {
        float xv = sh_x[m * 8 + k];
#pragma unroll
        for (int g = 0; g < 4; ++g) acc[g][r] += xv * wdec0r[g][k];
      }
    }
    mm_acc(acc, pDWhh0, sh_h0, wave, lane);
    __syncthreads();
    state_update(acc, c0, sh_h0, nullptr, wave, lane);
    __syncthreads();

    // dec layer 1
#pragma unroll
    for (int g = 0; g < 4; ++g)
#pragma unroll
      for (int r = 0; r < 8; ++r) acc[g][r] = bD1[g];
    mm_acc(acc, pDWih1, sh_h0, wave, lane);
    mm_acc(acc, pDWhh1, sh_h1, wave, lane);
    __syncthreads();
    state_update(acc, c1, sh_h1, sh_h1f, wave, lane);
    __syncthreads();

    // projection: pred = h1 @ outW^T + outB  (16x2, one wave)
    if (wave == 0) {
      int row = lane >> 1, col = lane & 1;
      float s = outB[col];
#pragma unroll 8
      for (int k = 0; k < HD; ++k) s += sh_h1f[row * SHF_STRIDE + k] * sh_outW[col * HD + k];
      out[((size_t)(m0 + row) * TDEC + t) * 2 + col] = s;
      sh_x[row * 8 + col] = s;            // feeds next step's x
    }
    __syncthreads();
  }
}

// ---------------------------------------------------------------------------
extern "C" void kernel_launch(void* const* d_in, const int* in_sizes, int n_in,
                              void* d_out, int out_size, void* d_ws, size_t ws_size,
                              hipStream_t stream) {
  const float* target  = (const float*)d_in[0];
  const float* encWih0 = (const float*)d_in[4];
  const float* encWhh0 = (const float*)d_in[5];
  const float* encBih0 = (const float*)d_in[6];
  const float* encBhh0 = (const float*)d_in[7];
  const float* encWih1 = (const float*)d_in[8];
  const float* encWhh1 = (const float*)d_in[9];
  const float* encBih1 = (const float*)d_in[10];
  const float* encBhh1 = (const float*)d_in[11];
  const float* decWih0 = (const float*)d_in[12];
  const float* decWhh0 = (const float*)d_in[13];
  const float* decBih0 = (const float*)d_in[14];
  const float* decBhh0 = (const float*)d_in[15];
  const float* decWih1 = (const float*)d_in[16];
  const float* decWhh1 = (const float*)d_in[17];
  const float* decBih1 = (const float*)d_in[18];
  const float* decBhh1 = (const float*)d_in[19];
  const float* outW    = (const float*)d_in[20];
  const float* outB    = (const float*)d_in[21];

  _Float16* packed = (_Float16*)d_ws;
  float* bsum = (float*)((char*)d_ws + (size_t)6 * PACK_MAT_ELEMS * sizeof(_Float16));

  pack_weights<<<(6 * PACK_MAT_ELEMS + 255) / 256, 256, 0, stream>>>(
      encWhh0, encWih1, encWhh1, decWhh0, decWih1, decWhh1, packed);
  pack_bias<<<(4 * NGATE + 255) / 256, 256, 0, stream>>>(
      encBih0, encBhh0, encBih1, encBhh1, decBih0, decBhh0, decBih1, decBhh1, bsum);

  int B = in_sizes[0] / (TENC * 7);
  lstm_seq2seq<<<B / 16, 256, 0, stream>>>(target, packed, bsum, encWih0, decWih0,
                                           outW, outB, (float*)d_out);
}
